// FlashNeoxAttention_67954972557443
// MI455X (gfx1250) — matrix operand order repaired and checked by reading, not compiled, not run on hardware
//
#include <hip/hip_runtime.h>
#include <stdint.h>

typedef __attribute__((ext_vector_type(16))) _Float16 v16h;
typedef __attribute__((ext_vector_type(8)))  float    v8f;

#define WMMA_F16(A_, B_, C_) \
  __builtin_amdgcn_wmma_f32_16x16x32_f16(false, (A_), false, (B_), (short)0, (C_), false, false)

// Async global->LDS copy of a 64B strip (4 x B128); INST_OFFSET applies to both
// the LDS and global addresses (ISA 10, 15.18.3). Tracked by ASYNCcnt.
#define ASYNC_TILE64B(ldsaddr_, gvoff_, sbase_) do {                                   \
  asm volatile("global_load_async_to_lds_b128 %0, %1, %2"                              \
               :: "v"(ldsaddr_), "v"(gvoff_), "s"(sbase_) : "memory");                 \
  asm volatile("global_load_async_to_lds_b128 %0, %1, %2 offset:16"                    \
               :: "v"(ldsaddr_), "v"(gvoff_), "s"(sbase_) : "memory");                 \
  asm volatile("global_load_async_to_lds_b128 %0, %1, %2 offset:32"                    \
               :: "v"(ldsaddr_), "v"(gvoff_), "s"(sbase_) : "memory");                 \
  asm volatile("global_load_async_to_lds_b128 %0, %1, %2 offset:48"                    \
               :: "v"(ldsaddr_), "v"(gvoff_), "s"(sbase_) : "memory");                 \
} while (0)

#define WAIT_ASYNC0() asm volatile("s_wait_asynccnt 0x0" ::: "memory")

// ---- WMMA fragment builders (contiguous k = 16*(lane/16) + s, wave32) ----
__device__ inline v16h frag_a(const _Float16* lds, int stride, int mbase, int kbase, int lane) {
  const int m = mbase + (lane & 15);
  const int khalf = (lane >> 4) << 4;
  v16h a;
#pragma unroll
  for (int v = 0; v < 8; ++v) {
    int k = kbase + khalf + (v << 1);
    a[2 * v]     = lds[m * stride + k];
    a[2 * v + 1] = lds[m * stride + k + 1];
  }
  return a;
}
__device__ inline v16h frag_b_kmajor(const _Float16* lds, int stride, int nbase, int kbase, int lane) {
  const int n = nbase + (lane & 15);
  const int ks = (lane >> 4) << 4;
  v16h b;
#pragma unroll
  for (int v = 0; v < 8; ++v) {
    int k = kbase + ks + 2 * v;
    b[2 * v]     = lds[k * stride + n];
    b[2 * v + 1] = lds[(k + 1) * stride + n];
  }
  return b;
}
__device__ inline v16h frag_b_nmajor(const _Float16* lds, int stride, int nbase, int kbase, int lane) {
  const int n = nbase + (lane & 15);
  const int ks = (lane >> 4) << 4;
  v16h b;
#pragma unroll
  for (int v = 0; v < 8; ++v) {
    int k = kbase + ks + 2 * v;
    b[2 * v]     = lds[n * stride + k];
    b[2 * v + 1] = lds[n * stride + k + 1];
  }
  return b;
}

// =====================================================================
// Kernel 1: qkv = X @ Wqkv^T + b, rotary on first 16 dims of q/k heads,
// scatter to q/k/v f16 buffers laid out [B,H,S,D].
// =====================================================================
__global__ __launch_bounds__(256) void qkv_rope_kernel(
    const float* __restrict__ X, const int* __restrict__ pos,
    const float* __restrict__ Wqkv, const float* __restrict__ bqkv,
    _Float16* __restrict__ qf, _Float16* __restrict__ kf, _Float16* __restrict__ vf) {
  __shared__ _Float16 Xs[64 * 36];
  __shared__ _Float16 Ws[64 * 36];
  const int tid = threadIdx.x, lane = tid & 31, w = tid >> 5;
  const int g  = blockIdx.x;          // 64-wide output group: (which,head)
  const int bm = blockIdx.y;          // 64-row token block
  const int mt = (w & 3) << 4;
  const int n0 = (w >> 2) << 5;
  const int row = tid >> 2, cb = (tid & 3) << 3;
  const int n = lane & 15, grp = lane >> 4;

  v8f acc0 = {}, acc1 = {};
  for (int k0 = 0; k0 < 2048; k0 += 32) {
    __syncthreads();
    const float* xs = X    + (size_t)(bm * 64 + row) * 2048 + k0 + cb;
    const float* ws = Wqkv + (size_t)(g  * 64 + row) * 2048 + k0 + cb;
#pragma unroll
    for (int i = 0; i < 8; ++i) {
      Xs[row * 36 + cb + i] = (_Float16)xs[i];
      Ws[row * 36 + cb + i] = (_Float16)ws[i];
    }
    __syncthreads();
    v16h a  = frag_a(Xs, 36, mt, 0, lane);
    v16h b0 = frag_b_nmajor(Ws, 36, n0, 0, lane);
    v16h b1 = frag_b_nmajor(Ws, 36, n0 + 16, 0, lane);
    acc0 = WMMA_F16(a, b0, acc0);
    acc1 = WMMA_F16(a, b1, acc1);
  }

  const float bias0 = bqkv[g * 64 + n0 + n];
  const float bias1 = bqkv[g * 64 + n0 + 16 + n];
#pragma unroll
  for (int v = 0; v < 8; ++v) { acc0[v] += bias0; acc1[v] += bias1; }

  if ((g < 64) && (n0 == 0)) {        // rotary on dims 0..15 of q/k heads
    const int j = n & 7;
    const float invf = __powf(10000.0f, -(float)j * 0.125f);
#pragma unroll
    for (int v = 0; v < 8; ++v) {
      int t = bm * 64 + mt + v + (grp << 3);
      float th = (float)pos[t] * invf;
      float c = __cosf(th), s = __sinf(th);
      float xs_ = acc0[v];
      float xo  = __shfl_xor(acc0[v], 8);
      acc0[v] = (n < 8) ? (xs_ * c - xo * s) : (xo * s + xs_ * c);
    }
  }

  const int which = g >> 5, h = g & 31;
  _Float16* dst = (which == 0) ? qf : (which == 1) ? kf : vf;
#pragma unroll
  for (int v = 0; v < 8; ++v) {
    int t = bm * 64 + mt + v + (grp << 3);
    int bb = t >> 10, ss = t & 1023;
    size_t base = (((size_t)(bb * 32 + h)) * 1024 + ss) * 64;
    dst[base + n0 + n]      = (_Float16)acc0[v];
    dst[base + n0 + 16 + n] = (_Float16)acc1[v];
  }
}

// =====================================================================
// Kernel 2: causal flash attention, one WG per (b,h,64-q-rows).
// 128 threads = 4 waves; double-buffered async K/V tiles (ASYNCcnt).
// =====================================================================
__global__ __launch_bounds__(128) void flash_attn_kernel(
    const _Float16* __restrict__ qf, const _Float16* __restrict__ kf,
    const _Float16* __restrict__ vf, _Float16* __restrict__ af) {
  __shared__ _Float16 Qs[64 * 72];
  __shared__ _Float16 Ks[2][64 * 72];
  __shared__ _Float16 Vs[2][64 * 72];
  __shared__ _Float16 Ps[4 * 16 * 72];
  const int tid = threadIdx.x, lane = tid & 31, w = tid >> 5;
  const int qb = blockIdx.x, h = blockIdx.y, b = blockIdx.z;
  const size_t head_base = ((size_t)(b * 32 + h)) * 1024 * 64;
  const int n = lane & 15, grp = lane >> 4;

  const _Float16* qbp = qf + head_base;
  const _Float16* kbp = kf + head_base;
  const _Float16* vbp = vf + head_base;

  // Per-thread strip: row r (0..63), 32 halfs (64B) at column c0.
  const int r = tid >> 1, c0 = (tid & 1) << 5;
  const unsigned strip = (unsigned)((r * 64 + c0) * 2);       // byte off in 64x64 tile
  const unsigned q_lds = (unsigned)(uintptr_t)&Qs[r * 72 + c0];

  // Prologue: async Q tile + K/V tile 0 into buffer 0.
  {
    unsigned qoff = (unsigned)(qb * 8192) + strip;
    ASYNC_TILE64B(q_lds, qoff, qbp);
    unsigned kl = (unsigned)(uintptr_t)&Ks[0][r * 72 + c0];
    unsigned vl = (unsigned)(uintptr_t)&Vs[0][r * 72 + c0];
    ASYNC_TILE64B(kl, strip, kbp);
    ASYNC_TILE64B(vl, strip, vbp);
  }
  WAIT_ASYNC0();
  __syncthreads();

  v16h qa0 = frag_a(Qs, 72, w << 4, 0, lane);
  v16h qa1 = frag_a(Qs, 72, w << 4, 32, lane);

  float row_max[8], row_sum[8];
  v8f o0 = {}, o1 = {}, o2 = {}, o3 = {};
#pragma unroll
  for (int v = 0; v < 8; ++v) { row_max[v] = -1e38f; row_sum[v] = 0.0f; }

  _Float16* Pw = Ps + w * 16 * 72;
  for (int kb = 0; kb <= qb; ++kb) {
    const int cur = kb & 1;
    if (kb > 0) {           // current tile was issued last iteration
      WAIT_ASYNC0();
      __syncthreads();      // also orders prior reads of buffer !cur before rewrite
    }
    if (kb + 1 <= qb) {     // prefetch next tile while computing this one
      unsigned off = (unsigned)((kb + 1) * 8192) + strip;
      unsigned kl = (unsigned)(uintptr_t)&Ks[cur ^ 1][r * 72 + c0];
      unsigned vl = (unsigned)(uintptr_t)&Vs[cur ^ 1][r * 72 + c0];
      ASYNC_TILE64B(kl, off, kbp);
      ASYNC_TILE64B(vl, off, vbp);
    }
    const _Float16* Kc = Ks[cur];
    const _Float16* Vc = Vs[cur];

    // S = Q @ K^T  (Kc stored [key][dim] => n-major for B)
    v8f s0 = {}, s1 = {}, s2 = {}, s3 = {};
    s0 = WMMA_F16(qa0, frag_b_nmajor(Kc, 72, 0, 0, lane), s0);
    s0 = WMMA_F16(qa1, frag_b_nmajor(Kc, 72, 0, 32, lane), s0);
    s1 = WMMA_F16(qa0, frag_b_nmajor(Kc, 72, 16, 0, lane), s1);
    s1 = WMMA_F16(qa1, frag_b_nmajor(Kc, 72, 16, 32, lane), s1);
    s2 = WMMA_F16(qa0, frag_b_nmajor(Kc, 72, 32, 0, lane), s2);
    s2 = WMMA_F16(qa1, frag_b_nmajor(Kc, 72, 32, 32, lane), s2);
    s3 = WMMA_F16(qa0, frag_b_nmajor(Kc, 72, 48, 0, lane), s3);
    s3 = WMMA_F16(qa1, frag_b_nmajor(Kc, 72, 48, 32, lane), s3);

    const bool diag = (kb == qb);
#pragma unroll
    for (int v = 0; v < 8; ++v) {
      int lm = v + (grp << 3);
      int m  = (w << 4) + lm;
      float sv0 = s0[v] * 0.125f, sv1 = s1[v] * 0.125f;
      float sv2 = s2[v] * 0.125f, sv3 = s3[v] * 0.125f;
      if (diag) {
        if (0  + n > m) sv0 = -1e30f;
        if (16 + n > m) sv1 = -1e30f;
        if (32 + n > m) sv2 = -1e30f;
        if (48 + n > m) sv3 = -1e30f;
      }
      float mx = fmaxf(fmaxf(sv0, sv1), fmaxf(sv2, sv3));
#pragma unroll
      for (int off = 1; off < 16; off <<= 1) mx = fmaxf(mx, __shfl_xor(mx, off));
      float nm = fmaxf(row_max[v], mx);
      float corr = __expf(row_max[v] - nm);
      row_max[v] = nm;
      float p0 = __expf(sv0 - nm), p1 = __expf(sv1 - nm);
      float p2 = __expf(sv2 - nm), p3 = __expf(sv3 - nm);
      float ps = p0 + p1 + p2 + p3;
#pragma unroll
      for (int off = 1; off < 16; off <<= 1) ps += __shfl_xor(ps, off);
      row_sum[v] = row_sum[v] * corr + ps;
      o0[v] *= corr; o1[v] *= corr; o2[v] *= corr; o3[v] *= corr;
      Pw[lm * 72 + 0  + n] = (_Float16)p0;
      Pw[lm * 72 + 16 + n] = (_Float16)p1;
      Pw[lm * 72 + 32 + n] = (_Float16)p2;
      Pw[lm * 72 + 48 + n] = (_Float16)p3;
    }
    // O += P @ V (per-wave LDS patch; DS ops in-order within a wave)
    v16h pa0 = frag_a(Pw, 72, 0, 0, lane);
    v16h pa1 = frag_a(Pw, 72, 0, 32, lane);
    o0 = WMMA_F16(pa0, frag_b_kmajor(Vc, 72, 0, 0, lane), o0);
    o0 = WMMA_F16(pa1, frag_b_kmajor(Vc, 72, 0, 32, lane), o0);
    o1 = WMMA_F16(pa0, frag_b_kmajor(Vc, 72, 16, 0, lane), o1);
    o1 = WMMA_F16(pa1, frag_b_kmajor(Vc, 72, 16, 32, lane), o1);
    o2 = WMMA_F16(pa0, frag_b_kmajor(Vc, 72, 32, 0, lane), o2);
    o2 = WMMA_F16(pa1, frag_b_kmajor(Vc, 72, 32, 32, lane), o2);
    o3 = WMMA_F16(pa0, frag_b_kmajor(Vc, 72, 48, 0, lane), o3);
    o3 = WMMA_F16(pa1, frag_b_kmajor(Vc, 72, 48, 32, lane), o3);
  }

#pragma unroll
  for (int v = 0; v < 8; ++v) {
    int lm = v + (grp << 3);
    int t = b * 1024 + qb * 64 + (w << 4) + lm;
    float inv = 1.0f / row_sum[v];
    size_t base = (size_t)t * 2048 + h * 64;
    af[base + 0  + n] = (_Float16)(o0[v] * inv);
    af[base + 16 + n] = (_Float16)(o1[v] * inv);
    af[base + 32 + n] = (_Float16)(o2[v] * inv);
    af[base + 48 + n] = (_Float16)(o3[v] * inv);
  }
}

// =====================================================================
// Kernel 3: out = attnO @ Wd^T + bd  (f32 output)
// =====================================================================
__global__ __launch_bounds__(256) void dense_kernel(
    const _Float16* __restrict__ A, const float* __restrict__ Wd,
    const float* __restrict__ bd, float* __restrict__ out) {
  __shared__ _Float16 Xs[64 * 36];
  __shared__ _Float16 Ws[64 * 36];
  const int tid = threadIdx.x, lane = tid & 31, w = tid >> 5;
  const int g = blockIdx.x, bm = blockIdx.y;
  const int mt = (w & 3) << 4, n0 = (w >> 2) << 5;
  const int row = tid >> 2, cb = (tid & 3) << 3;
  const int n = lane & 15, grp = lane >> 4;

  v8f acc0 = {}, acc1 = {};
  for (int k0 = 0; k0 < 2048; k0 += 32) {
    __syncthreads();
    const _Float16* xs = A  + (size_t)(bm * 64 + row) * 2048 + k0 + cb;
    const float*    ws = Wd + (size_t)(g  * 64 + row) * 2048 + k0 + cb;
#pragma unroll
    for (int i = 0; i < 8; ++i) {
      Xs[row * 36 + cb + i] = xs[i];
      Ws[row * 36 + cb + i] = (_Float16)ws[i];
    }
    __syncthreads();
    v16h a  = frag_a(Xs, 36, mt, 0, lane);
    v16h b0 = frag_b_nmajor(Ws, 36, n0, 0, lane);
    v16h b1 = frag_b_nmajor(Ws, 36, n0 + 16, 0, lane);
    acc0 = WMMA_F16(a, b0, acc0);
    acc1 = WMMA_F16(a, b1, acc1);
  }
  const float bias0 = bd[g * 64 + n0 + n];
  const float bias1 = bd[g * 64 + n0 + 16 + n];
#pragma unroll
  for (int v = 0; v < 8; ++v) {
    int t = bm * 64 + mt + v + (grp << 3);
    out[(size_t)t * 2048 + g * 64 + n0 + n]      = acc0[v] + bias0;
    out[(size_t)t * 2048 + g * 64 + n0 + 16 + n] = acc1[v] + bias1;
  }
}

extern "C" void kernel_launch(void* const* d_in, const int* in_sizes, int n_in,
                              void* d_out, int out_size, void* d_ws, size_t ws_size,
                              hipStream_t stream) {
  (void)in_sizes; (void)n_in; (void)out_size; (void)ws_size;
  const float* X    = (const float*)d_in[0];
  const int*   pos  = (const int*)d_in[1];
  const float* Wqkv = (const float*)d_in[4];
  const float* bqkv = (const float*)d_in[5];
  const float* Wd   = (const float*)d_in[6];
  const float* bd   = (const float*)d_in[7];
  float* out = (float*)d_out;

  const size_t HBSD = (size_t)4 * 32 * 1024 * 64;
  _Float16* qf = (_Float16*)d_ws;
  _Float16* kf = qf + HBSD;
  _Float16* vf = kf + HBSD;
  _Float16* af = vf + HBSD;

  qkv_rope_kernel<<<dim3(96, 64), 256, 0, stream>>>(X, pos, Wqkv, bqkv, qf, kf, vf);
  flash_attn_kernel<<<dim3(16, 32, 4), 128, 0, stream>>>(qf, kf, vf, af);
  dense_kernel<<<dim3(32, 64), 256, 0, stream>>>(af, Wd, bd, out);
}
